// RGCN_v1_45243185496739
// MI455X (gfx1250) — matrix operand order repaired and checked by reading, not compiled, not run on hardware
//
#include <hip/hip_runtime.h>
#include <hip/hip_bf16.h>

#define N_NODES 1024
#define SEQ     365
#define IN_DIM  16
#define HS      128

// LDS row strides padded so row-stride mod 64 banks == 4 words (conflict-free loads)
#define A_LD    1032   // 1024 + 8 bf16
#define H_LD    136    // 128 + 8 bf16

#define USE_ASYNC_LDS 1

typedef __bf16 bf16_t;
typedef bf16_t v8bf  __attribute__((ext_vector_type(8)));
typedef bf16_t v16bf __attribute__((ext_vector_type(16)));
typedef float  v8f   __attribute__((ext_vector_type(8)));

// fast activations: v_exp_f32 + v_rcp_f32, no IEEE divide sequences
__device__ __forceinline__ float sigmoid_f(float x) {
  return __builtin_amdgcn_rcpf(1.0f + __expf(-x));
}
__device__ __forceinline__ float tanh_f(float x) {
  const float e = __expf(-2.0f * x);
  return (1.0f - e) * __builtin_amdgcn_rcpf(1.0f + e);
}

__device__ __forceinline__ v8f splat8(float v) {
  v8f r;
#pragma unroll
  for (int i = 0; i < 8; ++i) r[i] = v;
  return r;
}

// ---- CDNA5 async global->LDS copy (ASYNCcnt path), 16B per lane ----
__device__ __forceinline__ void async_copy_b128(const bf16_t* __restrict__ g, bf16_t* l) {
#if USE_ASYNC_LDS
  asm volatile("global_load_async_to_lds_b128 %0, %1, off"
               :: "v"((unsigned)(size_t)l), "v"(g)
               : "memory");
#else
  *(v8bf*)l = *(const v8bf*)g;
#endif
}
__device__ __forceinline__ void async_join() {
#if USE_ASYNC_LDS
  asm volatile("s_wait_asynccnt 0" ::: "memory");
#endif
  __syncthreads();
}

// A-matrix (MxK, row-major bf16) fragment for V_WMMA_*_16X16X32_BF16.
// lanes 0-15: m=lane,    K = {kc+0..7, kc+16..23}
// lanes16-31: m=lane-16, K = {kc+8..15, kc+24..31}
__device__ __forceinline__ v16bf frag_a_rm(const bf16_t* __restrict__ base, int ld,
                                           int lane, int kc) {
  const int m  = lane & 15;
  const int k0 = kc + ((lane & 16) ? 8 : 0);
  const bf16_t* p = base + (size_t)m * ld + k0;
  v8bf lo = *(const v8bf*)(p);
  v8bf hi = *(const v8bf*)(p + 16);
  return __builtin_shufflevector(lo, hi, 0,1,2,3,4,5,6,7,8,9,10,11,12,13,14,15);
}

// B-matrix (KxN) fragment from a transposed [N][K] row-major bf16 array.
__device__ __forceinline__ v16bf frag_b_t(const bf16_t* __restrict__ bt, int ld,
                                          int lane, int kc, int nbase) {
  const int n  = nbase + (lane & 15);
  const int k0 = kc + ((lane & 16) ? 8 : 0);
  const bf16_t* p = bt + (size_t)n * ld + k0;
  v8bf lo = *(const v8bf*)(p);
  v8bf hi = *(const v8bf*)(p + 16);
  return __builtin_shufflevector(lo, hi, 0,1,2,3,4,5,6,7,8,9,10,11,12,13,14,15);
}

// x_t as A-matrix with K padded 16 -> 32 (upper half zero), f32 -> bf16 on the fly.
__device__ __forceinline__ v16bf frag_a_x(const float* __restrict__ xblk, int lane, int t) {
  const int m    = lane & 15;
  const int koff = (lane & 16) ? 8 : 0;
  const float* p = xblk + ((size_t)m * SEQ + t) * IN_DIM + koff;
  v16bf r;
#pragma unroll
  for (int i = 0; i < 8; ++i) r[i] = (bf16_t)p[i];
#pragma unroll
  for (int i = 0; i < 8; ++i) r[8 + i] = (bf16_t)0.0f;
  return r;
}

__device__ __forceinline__ v8f wmma_bf16(v16bf a, v16bf b, v8f c) {
  return __builtin_amdgcn_wmma_f32_16x16x32_bf16(false, a, false, b, (short)0, c,
                                                 false, false);
}

// ---------------- prep: bf16 conversions, weight transposes, state zeroing ------------
__global__ __launch_bounds__(256) void rgcn_prep_kernel(
    const float* __restrict__ A, const float* __restrict__ w_ih,
    const float* __restrict__ w_hh, const float* __restrict__ w_q,
    bf16_t* __restrict__ a_bf, bf16_t* __restrict__ wih_t,
    bf16_t* __restrict__ whh_t, bf16_t* __restrict__ wq_t,
    bf16_t* __restrict__ h_bf, float* __restrict__ c_st) {
  const int tid = blockIdx.x * blockDim.x + threadIdx.x;
  const int nt  = gridDim.x * blockDim.x;
  for (int i = tid; i < N_NODES * N_NODES; i += nt) a_bf[i] = (bf16_t)A[i];
  for (int i = tid; i < 4 * HS * HS; i += nt) {            // whh_t[n][k] = w_hh[k][n]
    int n = i / HS, k = i % HS;
    whh_t[i] = (bf16_t)w_hh[k * (4 * HS) + n];
  }
  for (int i = tid; i < HS * HS; i += nt) {                // wq_t[n][k] = w_q[k][n]
    int n = i / HS, k = i % HS;
    wq_t[i] = (bf16_t)w_q[k * HS + n];
  }
  for (int i = tid; i < 4 * HS * 32; i += nt) {            // wih_t[n][k], K padded to 32
    int n = i / 32, k = i % 32;
    wih_t[i] = (k < IN_DIM) ? (bf16_t)w_ih[k * (4 * HS) + n] : (bf16_t)0.0f;
  }
  for (int i = tid; i < N_NODES * HS; i += nt) {           // h0 = c0 = 0
    h_bf[i] = (bf16_t)0.0f;
    c_st[i] = 0.0f;
  }
}

// ---------------- per step, phase 1: Qt = tanh(h @ W_q + b_q), stored [hs][node] ------
__global__ __launch_bounds__(256) void rgcn_q_kernel(
    const bf16_t* __restrict__ h_bf, const bf16_t* __restrict__ wq_t,
    const float* __restrict__ bias_q, bf16_t* __restrict__ qt) {
  __shared__ __align__(16) bf16_t h_sh[16 * H_LD];
  const int wave  = threadIdx.x >> 5;
  const int lane  = threadIdx.x & 31;
  const int mbase = blockIdx.x * 16;
  const bf16_t* hrow = h_bf + (size_t)mbase * HS;
  const int n = wave * 16 + (lane & 15);

  // stage h tile (16 x 128 bf16 = 4KB) via async copy, shared by all 8 waves
  {
    const int r = threadIdx.x >> 4, c = threadIdx.x & 15;
    async_copy_b128(hrow + r * HS + c * 8, h_sh + r * H_LD + c * 8);
  }
  async_join();

  // two independent WMMA chains; bias folded into accumulator init
  v8f acc0 = splat8(bias_q[n]), acc1 = 0.0f;
#pragma unroll
  for (int kc = 0; kc < HS; kc += 64) {
    acc0 = wmma_bf16(frag_a_rm(h_sh, H_LD, lane, kc),
                     frag_b_t(wq_t, HS, lane, kc, wave * 16), acc0);
    acc1 = wmma_bf16(frag_a_rm(h_sh, H_LD, lane, kc + 32),
                     frag_b_t(wq_t, HS, lane, kc + 32, wave * 16), acc1);
  }
  v8f acc = acc0 + acc1;

  const int m0 = (lane & 16) ? 8 : 0;
  v8bf qv;
#pragma unroll
  for (int r = 0; r < 8; ++r) qv[r] = (bf16_t)tanh_f(acc[r]);
  *(v8bf*)(qt + (size_t)n * N_NODES + mbase + m0) = qv;    // one 16B store/lane
}

// ---------------- per step, phase 2: AQ + gates + cell update + output ----------------
__global__ __launch_bounds__(256) void rgcn_step_kernel(
    const float* __restrict__ x, const bf16_t* __restrict__ a_bf,
    const bf16_t* __restrict__ qt, const bf16_t* __restrict__ whh_t,
    const bf16_t* __restrict__ wih_t, const float* __restrict__ bias,
    const float* __restrict__ dense_w, const float* __restrict__ dense_b,
    bf16_t* __restrict__ h_bf, float* __restrict__ c_st,
    float* __restrict__ out, int t) {
  __shared__ __align__(16) bf16_t a_sh[16 * A_LD];   // 16 x 1024 A-slice (+pad) ~33KB
  __shared__ __align__(16) bf16_t h_sh[16 * H_LD];   // 16 x 128 h(t-1) tile
  __shared__ float h_lds[16][HS + 1];                // h(t) for output projection
  const int wave  = threadIdx.x >> 5;
  const int lane  = threadIdx.x & 31;
  const int mbase = blockIdx.x * 16;
  const int hs    = wave * 16 + (lane & 15);

  // ---- async-stage A row-slice (32KB) and h tile (4KB) into LDS ----
  {
    const bf16_t* arow = a_bf + (size_t)mbase * N_NODES;
    for (int i = threadIdx.x; i < 2048; i += 256) {        // 2048 x 16B = 32KB
      const int r = i >> 7, c = i & 127;
      async_copy_b128(arow + r * N_NODES + c * 8, a_sh + r * A_LD + c * 8);
    }
    const bf16_t* hrow = h_bf + (size_t)mbase * HS;
    const int r = threadIdx.x >> 4, c = threadIdx.x & 15;
    async_copy_b128(hrow + r * HS + c * 8, h_sh + r * H_LD + c * 8);
  }
  async_join();

  // ---- AQ tile: rows [mbase,mbase+16), hs-cols [16w,16w+16), K = 1024 nodes ----
  // two independent accumulator chains to pipeline the XDL
  v8f aq0 = 0.0f, aq1 = 0.0f;
#pragma unroll 4
  for (int kc = 0; kc < N_NODES; kc += 64) {
    aq0 = wmma_bf16(frag_a_rm(a_sh, A_LD, lane, kc),
                    frag_b_t(qt, N_NODES, lane, kc, wave * 16), aq0);
    aq1 = wmma_bf16(frag_a_rm(a_sh, A_LD, lane, kc + 32),
                    frag_b_t(qt, N_NODES, lane, kc + 32, wave * 16), aq1);
  }
  v8f aq = aq0 + aq1;

  // ---- gate tiles i/f/g/o: 4 independent chains, bias pre-loaded into accumulators --
  v8f gi = splat8(bias[hs]);
  v8f gf = splat8(bias[HS + hs]);
  v8f gg = splat8(bias[2 * HS + hs]);
  v8f go = splat8(bias[3 * HS + hs]);
#pragma unroll
  for (int kc = 0; kc < HS; kc += 32) {
    v16bf a = frag_a_rm(h_sh, H_LD, lane, kc);
    gi = wmma_bf16(a, frag_b_t(whh_t, HS, lane, kc, 0 * HS + wave * 16), gi);
    gf = wmma_bf16(a, frag_b_t(whh_t, HS, lane, kc, 1 * HS + wave * 16), gf);
    gg = wmma_bf16(a, frag_b_t(whh_t, HS, lane, kc, 2 * HS + wave * 16), gg);
    go = wmma_bf16(a, frag_b_t(whh_t, HS, lane, kc, 3 * HS + wave * 16), go);
  }
  {
    v16bf ax = frag_a_x(x + (size_t)mbase * SEQ * IN_DIM, lane, t);
    gi = wmma_bf16(ax, frag_b_t(wih_t, 32, lane, 0, 0 * HS + wave * 16), gi);
    gf = wmma_bf16(ax, frag_b_t(wih_t, 32, lane, 0, 1 * HS + wave * 16), gf);
    gg = wmma_bf16(ax, frag_b_t(wih_t, 32, lane, 0, 2 * HS + wave * 16), gg);
    go = wmma_bf16(ax, frag_b_t(wih_t, 32, lane, 0, 3 * HS + wave * 16), go);
  }

  // ---- elementwise LSTM cell update; element (m = r + 8*(lane>=16), hs = 16w+lane&15)
  // (gates were read from the LDS copy of h, so rewriting h_bf below is hazard-free)
  const int m0 = (lane & 16) ? 8 : 0;
#pragma unroll
  for (int r = 0; r < 8; ++r) {
    const size_t idx = (size_t)(mbase + m0 + r) * HS + hs;
    const float it = sigmoid_f(gi[r]);
    const float ft = sigmoid_f(gf[r]);
    const float gt = tanh_f(gg[r]);
    const float ot = sigmoid_f(go[r]);
    const float cn = ft * (c_st[idx] + aq[r]) + it * gt;
    const float hv = ot * tanh_f(cn);
    c_st[idx] = cn;
    h_bf[idx] = (bf16_t)hv;
    h_lds[m0 + r][hs] = hv;
  }
  __syncthreads();

  // ---- out[node][t] = h . dense_w + dense_b : 8 lanes per node + shfl butterfly ----
  if (threadIdx.x < 128) {
    const int m = threadIdx.x >> 3;
    const int j = threadIdx.x & 7;
    float s = 0.0f;
#pragma unroll
    for (int i = 0; i < 16; ++i) s += h_lds[m][j + 8 * i] * dense_w[j + 8 * i];
    s += __shfl_xor(s, 1, 32);
    s += __shfl_xor(s, 2, 32);
    s += __shfl_xor(s, 4, 32);
    if (j == 0) out[(size_t)(mbase + m) * SEQ + t] = s + dense_b[0];
  }
}

extern "C" void kernel_launch(void* const* d_in, const int* in_sizes, int n_in,
                              void* d_out, int out_size, void* d_ws, size_t ws_size,
                              hipStream_t stream) {
  const float* x     = (const float*)d_in[0];
  const float* A     = (const float*)d_in[1];
  const float* w_ih  = (const float*)d_in[2];
  const float* w_hh  = (const float*)d_in[3];
  const float* bias  = (const float*)d_in[4];
  const float* w_q   = (const float*)d_in[5];
  const float* b_q   = (const float*)d_in[6];
  const float* d_w   = (const float*)d_in[7];
  const float* d_b   = (const float*)d_in[8];
  float* out = (float*)d_out;

  char* ws = (char*)d_ws;
  size_t off = 0;
  bf16_t* a_bf  = (bf16_t*)(ws + off); off += (size_t)N_NODES * N_NODES * 2;  // 2 MB
  bf16_t* qt    = (bf16_t*)(ws + off); off += (size_t)HS * N_NODES * 2;       // 256 KB
  bf16_t* h_bf  = (bf16_t*)(ws + off); off += (size_t)N_NODES * HS * 2;       // 256 KB
  float*  c_st  = (float*) (ws + off); off += (size_t)N_NODES * HS * 4;       // 512 KB
  bf16_t* whh_t = (bf16_t*)(ws + off); off += (size_t)4 * HS * HS * 2;        // 128 KB
  bf16_t* wq_t  = (bf16_t*)(ws + off); off += (size_t)HS * HS * 2;            // 32 KB
  bf16_t* wih_t = (bf16_t*)(ws + off); off += (size_t)4 * HS * 32 * 2;        // 32 KB

  rgcn_prep_kernel<<<512, 256, 0, stream>>>(A, w_ih, w_hh, w_q, a_bf, wih_t,
                                            whh_t, wq_t, h_bf, c_st);
  for (int t = 0; t < SEQ; ++t) {
    rgcn_q_kernel<<<N_NODES / 16, 256, 0, stream>>>(h_bf, wq_t, b_q, qt);
    rgcn_step_kernel<<<N_NODES / 16, 256, 0, stream>>>(
        x, a_bf, qt, whh_t, wih_t, bias, d_w, d_b, h_bf, c_st, out, t);
  }
}
